// TransformerDetector_28853590294727
// MI455X (gfx1250) — matrix operand order repaired and checked by reading, not compiled, not run on hardware
//
#include <hip/hip_runtime.h>

typedef _Float16 v8h  __attribute__((ext_vector_type(8)));
typedef _Float16 v16h __attribute__((ext_vector_type(16)));
typedef float    v8f  __attribute__((ext_vector_type(8)));

#define BM 128
#define BN 64
#define BKC 32
#define LDST (BKC + 8)   // 40 halfs = 80B row stride, multiple of 16B

#if defined(__has_builtin)
#if __has_builtin(__builtin_amdgcn_sched_barrier)
#define SCHED_FENCE() __builtin_amdgcn_sched_barrier(0)
#endif
#endif
#ifndef SCHED_FENCE
#define SCHED_FENCE()
#endif

// ---------------------------------------------------------------------------
// Tiled WMMA GEMM with double-buffered LDS staging.
//   C = act(scale*(A@B) + bias) + resid
//   NT (TRANSB=false): B is W[N,K] row-major (torch Linear weight)
//   NN (TRANSB=true):  B is  [K,N] row-major (attn V)
// Batched over blockIdx.z with two nested strides (z1 = z/nz2, z2 = z%nz2).
// Rows of A beyond aRows and rows of B beyond bRows read as zero (padding).
// ---------------------------------------------------------------------------
template<bool TRANSB>
__global__ __launch_bounds__(256) void wmma_gemm_kernel(
    const float* __restrict__ A, int lda, int aRows, long aS1, long aS2,
    const float* __restrict__ B, int ldb, int bRows, long bS1, long bS2,
    const float* __restrict__ bias, const float* __restrict__ resid,
    float* __restrict__ C, int ldc, long cS1, long cS2,
    int Kdim, int mStore, int nStore, int nz2, float scale, int relu)
{
    __shared__ __align__(16) _Float16 As[2][BM][LDST];
    __shared__ __align__(16) _Float16 Bs[2][BN][LDST];

    const int tid  = threadIdx.x;
    const int wv   = tid >> 5;
    const int lane = tid & 31;
    const int l15  = lane & 15;
    const int hi16 = lane >> 4;                 // 0 or 1

    const int mBase = blockIdx.x * BM;
    const int nBase = blockIdx.y * BN;
    const int z1 = (int)blockIdx.z / nz2;
    const int z2 = (int)blockIdx.z % nz2;
    const float* Ab = A + (long)z1 * aS1 + (long)z2 * aS2;
    const float* Bb = B + (long)z1 * bS1 + (long)z2 * bS2;
    float*       Cb = C + (long)z1 * cS1 + (long)z2 * cS2;
    const float* Rb = resid ? (resid + (long)z1 * cS1 + (long)z2 * cS2) : nullptr;

    v8f acc[4];
#pragma unroll
    for (int t = 0; t < 4; ++t)
#pragma unroll
        for (int i = 0; i < 8; ++i) acc[t][i] = 0.0f;

    const int ar  = tid >> 1;            // A stage: 128 rows x 2 half-rows
    const int ac  = (tid & 1) * 16;
    const int bnr = tid >> 2;            // B NT stage: 64 rows x 4 chunks
    const int bnc = (tid & 3) * 8;
    const int bkk = tid & 31;            // B NN stage: 32 k-rows x 8 cols
    const int bn0 = (tid >> 5) * 8;

    // ---- stagers: global f32 -> f16 in LDS buffer `buf`, K-chunk at `kc` ----
    auto stageA = [&](int buf, int kc) {
        const int grow = mBase + ar;
        if (grow < aRows) {
            const float4* src =
                reinterpret_cast<const float4*>(Ab + (long)grow * lda + kc + ac);
#pragma unroll
            for (int j = 0; j < 4; ++j) {
                float4 v = src[j];
                As[buf][ar][ac + 4*j + 0] = (_Float16)v.x;
                As[buf][ar][ac + 4*j + 1] = (_Float16)v.y;
                As[buf][ar][ac + 4*j + 2] = (_Float16)v.z;
                As[buf][ar][ac + 4*j + 3] = (_Float16)v.w;
            }
        } else {
#pragma unroll
            for (int j = 0; j < 16; ++j) As[buf][ar][ac + j] = (_Float16)0.0f;
        }
    };
    auto stageB = [&](int buf, int kc) {
        if (!TRANSB) {
            const int gn = nBase + bnr;
            if (gn < bRows) {
                const float4* src =
                    reinterpret_cast<const float4*>(Bb + (long)gn * ldb + kc + bnc);
#pragma unroll
                for (int j = 0; j < 2; ++j) {
                    float4 v = src[j];
                    Bs[buf][bnr][bnc + 4*j + 0] = (_Float16)v.x;
                    Bs[buf][bnr][bnc + 4*j + 1] = (_Float16)v.y;
                    Bs[buf][bnr][bnc + 4*j + 2] = (_Float16)v.z;
                    Bs[buf][bnr][bnc + 4*j + 3] = (_Float16)v.w;
                }
            } else {
#pragma unroll
                for (int j = 0; j < 8; ++j) Bs[buf][bnr][bnc + j] = (_Float16)0.0f;
            }
        } else {
            const int gk = kc + bkk;
            if (gk < bRows) {
                const float4* src =
                    reinterpret_cast<const float4*>(Bb + (long)gk * ldb + nBase + bn0);
                float4 v0 = src[0], v1 = src[1];
                Bs[buf][bn0 + 0][bkk] = (_Float16)v0.x;
                Bs[buf][bn0 + 1][bkk] = (_Float16)v0.y;
                Bs[buf][bn0 + 2][bkk] = (_Float16)v0.z;
                Bs[buf][bn0 + 3][bkk] = (_Float16)v0.w;
                Bs[buf][bn0 + 4][bkk] = (_Float16)v1.x;
                Bs[buf][bn0 + 5][bkk] = (_Float16)v1.y;
                Bs[buf][bn0 + 6][bkk] = (_Float16)v1.z;
                Bs[buf][bn0 + 7][bkk] = (_Float16)v1.w;
            } else {
#pragma unroll
                for (int j = 0; j < 8; ++j) Bs[buf][bn0 + j][bkk] = (_Float16)0.0f;
            }
        }
    };

    // prologue: fill buffer 0
    stageA(0, 0);
    stageB(0, 0);
    __syncthreads();

    int cur = 0;
    for (int kc = 0; kc < Kdim; kc += BKC) {
        const int nxt = cur ^ 1;
        // issue next tile's global loads + LDS stores (no barrier needed:
        // they target the other buffer) — overlaps with this tile's WMMAs
        if (kc + BKC < Kdim) {
            stageA(nxt, kc + BKC);
            stageB(nxt, kc + BKC);
        }

        // ---- A fragment (ISA 16-bit A layout: lanes 0-15 M, K split 0-7/8-15
        //      across lane halves; K 16-31 in upper 4 VGPRs) ----
        const _Float16* ap = &As[cur][wv * 16 + l15][0];
        const int akb = hi16 * 8;
        v8h alo = *(const v8h*)(ap + akb);
        v8h ahi = *(const v8h*)(ap + 16 + akb);
        v16h af;
#pragma unroll
        for (int i = 0; i < 8; ++i) { af[i] = alo[i]; af[i + 8] = ahi[i]; }

        // preload ALL 4 B fragments, then fence the scheduler so the four
        // WMMAs issue back-to-back behind a single s_wait_dscnt
        v16h bfrag[4];
#pragma unroll
        for (int t = 0; t < 4; ++t) {
            const _Float16* bp = &Bs[cur][t * 16 + l15][0];
            const int bkb = hi16 * 8;
            v8h blo = *(const v8h*)(bp + bkb);
            v8h bhi = *(const v8h*)(bp + 16 + bkb);
#pragma unroll
            for (int i = 0; i < 8; ++i) { bfrag[t][i] = blo[i]; bfrag[t][i + 8] = bhi[i]; }
        }
        SCHED_FENCE();
#pragma unroll
        for (int t = 0; t < 4; ++t)
            acc[t] = __builtin_amdgcn_wmma_f32_16x16x32_f16(
                false, af, false, bfrag[t], (short)0, acc[t], false, false);
        SCHED_FENCE();

        __syncthreads();   // nxt stores visible; cur free for overwrite
        cur = nxt;
    }

    // ---- epilogue: C/D layout — VGPR j: M = j + 8*hi16, N = l15 ----
    const int mrow = mBase + wv * 16 + hi16 * 8;
#pragma unroll
    for (int t = 0; t < 4; ++t) {
        const int n = nBase + t * 16 + l15;
        if (n >= nStore) continue;
        const float bv = bias ? bias[n] : 0.0f;
#pragma unroll
        for (int j = 0; j < 8; ++j) {
            const int m = mrow + j;
            if (m >= mStore) continue;
            float v = acc[t][j] * scale + bv;
            if (relu) v = fmaxf(v, 0.0f);
            if (Rb) v += Rb[(long)m * ldc + n];
            Cb[(long)m * ldc + n] = v;
        }
    }
}

// ---------------------------------------------------------------------------
// LayerNorm over D=256, one block per row.
// ---------------------------------------------------------------------------
__global__ __launch_bounds__(256) void layernorm_kernel(
    const float* __restrict__ x, const float* __restrict__ g,
    const float* __restrict__ b, float* __restrict__ out, int D)
{
    __shared__ float red[256];
    const long row = blockIdx.x;
    const int tid = threadIdx.x;
    float v = x[row * D + tid];
    red[tid] = v; __syncthreads();
    for (int s = 128; s > 0; s >>= 1) {
        if (tid < s) red[tid] += red[tid + s];
        __syncthreads();
    }
    float mean = red[0] / (float)D; __syncthreads();
    float d = v - mean;
    red[tid] = d * d; __syncthreads();
    for (int s = 128; s > 0; s >>= 1) {
        if (tid < s) red[tid] += red[tid + s];
        __syncthreads();
    }
    float var = red[0] / (float)D;
    out[row * D + tid] = d * rsqrtf(var + 1e-5f) * g[tid] + b[tid];
}

// ---------------------------------------------------------------------------
// Row softmax over ld=512 columns, first `valid` are real, rest forced to 0
// (so the zero-padded AV K-loop is exact).
// ---------------------------------------------------------------------------
__global__ __launch_bounds__(512) void softmax_kernel(
    float* __restrict__ S, int ld, int valid)
{
    __shared__ float red[512];
    const long row = blockIdx.x;
    const int tid = threadIdx.x;
    float* r = S + row * (long)ld;
    float v = (tid < valid) ? r[tid] : -3.4e38f;
    red[tid] = v; __syncthreads();
    for (int s = 256; s > 0; s >>= 1) {
        if (tid < s) red[tid] = fmaxf(red[tid], red[tid + s]);
        __syncthreads();
    }
    float mx = red[0]; __syncthreads();
    float e = (tid < valid) ? expf(v - mx) : 0.0f;
    red[tid] = e; __syncthreads();
    for (int s = 256; s > 0; s >>= 1) {
        if (tid < s) red[tid] += red[tid + s];
        __syncthreads();
    }
    r[tid] = e / red[0];
}

// sd_in[(b*2+a)*500+t][0:512]=feats[b][t], [512:704]=anchors[b][a]
__global__ void concat_kernel(const float* __restrict__ feats,
                              const float* __restrict__ anchors,
                              float* __restrict__ out, long total)
{
    long i = blockIdx.x * (long)blockDim.x + threadIdx.x;
    if (i >= total) return;
    int  c   = (int)(i % 704);
    long tok = i / 704;
    int  t   = (int)(tok % 500);
    long ba  = tok / 500;
    int  a   = (int)(ba % 2);
    int  b   = (int)(ba / 2);
    out[i] = (c < 512) ? feats[((long)b * 500 + t) * 512 + c]
                       : anchors[((long)b * 2 + a) * 192 + (c - 512)];
}

// comb_in[(b*500+t)*512 + a*256 + c] = sd[((b*2+a)*500+t)*256 + c] + PE(t,c)
__global__ void posenc_transpose_kernel(const float* __restrict__ sd,
                                        float* __restrict__ out, long total)
{
    long i = blockIdx.x * (long)blockDim.x + threadIdx.x;
    if (i >= total) return;
    int  c512 = (int)(i % 512);
    long bt   = i / 512;
    int  t    = (int)(bt % 500);
    int  b    = (int)(bt / 500);
    int  a    = c512 >> 8;
    int  c    = c512 & 255;
    float ex  = (float)(c & ~1) / 256.0f;          // 2*(c//2)/d
    float ang = (float)t * powf(10000.0f, -ex);
    float pe  = (c & 1) ? cosf(ang) : sinf(ang);
    out[i] = sd[(((long)b * 2 + a) * 500 + t) * 256 + c] + pe;
}

// out[(b*500+t)*2 + a] = dot(x[(b*500+t)*256 + a*128 .. +128], Wout) + bout
__global__ void out_proj_kernel(const float* __restrict__ x,
                                const float* __restrict__ w,
                                const float* __restrict__ b,
                                float* __restrict__ out, int total)
{
    int i = blockIdx.x * blockDim.x + threadIdx.x;
    if (i >= total) return;
    int  a   = i & 1;
    long row = i >> 1;
    const float* p = x + row * 256 + a * 128;
    float s = b[0];
#pragma unroll 8
    for (int j = 0; j < 128; ++j) s += p[j] * w[j];
    out[i] = s;
}

// ---------------------------------------------------------------------------
// Host-side orchestration
// ---------------------------------------------------------------------------
struct LayerP {
    const float *W1,*W2,*Wk,*Wo,*Wq,*Wv,*b1,*b2,*bf1,*bf2,*bk,*bo,*bq,*bv,*g1,*g2;
};
struct EncParams { const float *Win,*b_out,*bin,*g_out; LayerP l[2]; };
struct Bufs { float *A0,*A1,*Q,*K,*V,*ATT,*F1,*SC; };

static EncParams load_enc(void* const* din, int base)
{
    auto F = [&](int i) { return (const float*)din[i]; };
    EncParams e;
    e.Win = F(base + 0); e.b_out = F(base + 1);
    e.bin = F(base + 2); e.g_out = F(base + 3);
    for (int l = 0; l < 2; ++l) {
        int o = base + 4 + l * 16;   // sorted: W1 W2 Wk Wo Wq Wv b1 b2 bf1 bf2 bk bo bq bv g1 g2
        e.l[l] = { F(o+0),F(o+1),F(o+2),F(o+3),F(o+4),F(o+5),F(o+6),F(o+7),
                   F(o+8),F(o+9),F(o+10),F(o+11),F(o+12),F(o+13),F(o+14),F(o+15) };
    }
    return e;
}

static inline void gemm(bool transb,
    const float* A, int lda, int aRows, long aS1, long aS2,
    const float* B, int ldb, int bRows, long bS1, long bS2,
    const float* bias, const float* resid,
    float* C, int ldc, long cS1, long cS2,
    int M, int N, int K, int mStore, int nStore, int z, int nz2,
    float scale, int relu, hipStream_t st)
{
    dim3 g(M / BM, N / BN, z), b(256);
    if (transb)
        wmma_gemm_kernel<true><<<g, b, 0, st>>>(A, lda, aRows, aS1, aS2,
            B, ldb, bRows, bS1, bS2, bias, resid, C, ldc, cS1, cS2,
            K, mStore, nStore, nz2, scale, relu);
    else
        wmma_gemm_kernel<false><<<g, b, 0, st>>>(A, lda, aRows, aS1, aS2,
            B, ldb, bRows, bS1, bS2, bias, resid, C, ldc, cS1, cS2,
            K, mStore, nStore, nz2, scale, relu);
}

static void run_encoder(const float* x, int idim, int tokens, int seqs,
                        const EncParams& ep, const Bufs& bf, hipStream_t st)
{
    const int T = 500, Tp = 512, H = 4, D = 256, E = 512, CH = 8;

    // input projection
    gemm(false, x, idim, tokens, 0, 0, ep.Win, idim, D, 0, 0, ep.bin, nullptr,
         bf.A0, D, 0, 0, tokens, D, idim, tokens, D, 1, 1, 1.0f, 0, st);

    for (int l = 0; l < 2; ++l) {
        const LayerP& L = ep.l[l];
        layernorm_kernel<<<tokens, 256, 0, st>>>(bf.A0, L.g1, L.b1, bf.A1, D);

        gemm(false, bf.A1, D, tokens, 0, 0, L.Wq, D, D, 0, 0, L.bq, nullptr,
             bf.Q, D, 0, 0, tokens, D, D, tokens, D, 1, 1, 1.0f, 0, st);
        gemm(false, bf.A1, D, tokens, 0, 0, L.Wk, D, D, 0, 0, L.bk, nullptr,
             bf.K, D, 0, 0, tokens, D, D, tokens, D, 1, 1, 1.0f, 0, st);
        gemm(false, bf.A1, D, tokens, 0, 0, L.Wv, D, D, 0, 0, L.bv, nullptr,
             bf.V, D, 0, 0, tokens, D, D, tokens, D, 1, 1, 1.0f, 0, st);

        for (int s0 = 0; s0 < seqs; s0 += CH) {
            // scores = Q @ K^T / 8   (batched over z = (seq, head))
            gemm(false, bf.Q + (long)s0 * T * D, D, T, (long)T * D, 64,
                        bf.K + (long)s0 * T * D, D, T, (long)T * D, 64,
                 nullptr, nullptr, bf.SC, Tp, (long)H * Tp * Tp, (long)Tp * Tp,
                 Tp, Tp, 64, Tp, Tp, CH * H, H, 0.125f, 0, st);
            softmax_kernel<<<CH * H * Tp, Tp, 0, st>>>(bf.SC, Tp, T);
            // out = att @ V  (B in K-major layout -> TRANSB)
            gemm(true, bf.SC, Tp, Tp, (long)H * Tp * Tp, (long)Tp * Tp,
                       bf.V + (long)s0 * T * D, D, T, (long)T * D, 64,
                 nullptr, nullptr, bf.ATT + (long)s0 * T * D, D, (long)T * D, 64,
                 Tp, 64, Tp, T, 64, CH * H, H, 1.0f, 0, st);
        }

        // h = ln1(e) + attn_out @ Wo^T + bo
        gemm(false, bf.ATT, D, tokens, 0, 0, L.Wo, D, D, 0, 0, L.bo, bf.A1,
             bf.A0, D, 0, 0, tokens, D, D, tokens, D, 1, 1, 1.0f, 0, st);
        layernorm_kernel<<<tokens, 256, 0, st>>>(bf.A0, L.g2, L.b2, bf.A1, D);
        // FFN
        gemm(false, bf.A1, D, tokens, 0, 0, L.W1, D, E, 0, 0, L.bf1, nullptr,
             bf.F1, E, 0, 0, tokens, E, D, tokens, E, 1, 1, 1.0f, 1, st);
        gemm(false, bf.F1, E, tokens, 0, 0, L.W2, E, D, 0, 0, L.bf2, bf.A1,
             bf.A0, D, 0, 0, tokens, D, E, tokens, D, 1, 1, 1.0f, 0, st);
    }
    layernorm_kernel<<<tokens, 256, 0, st>>>(bf.A0, ep.g_out, ep.b_out, bf.A1, D);
}

extern "C" void kernel_launch(void* const* d_in, const int* in_sizes, int n_in,
                              void* d_out, int out_size, void* d_ws, size_t ws_size,
                              hipStream_t stream)
{
    (void)in_sizes; (void)n_in; (void)out_size; (void)ws_size;

    // JAX pytree (sorted-dict) input order:
    // 0 feats, 1 anchors, 2 Wout, 3 bout, 4.. comb enc (36), 40.. det enc (36)
    const float* feats   = (const float*)d_in[0];
    const float* anchors = (const float*)d_in[1];
    const float* Wout    = (const float*)d_in[2];
    const float* bout    = (const float*)d_in[3];
    EncParams comb = load_enc(d_in, 4);
    EncParams det  = load_enc(d_in, 40);

    char* p = (char*)d_ws;
    auto alloc = [&](size_t bytes) { float* r = (float*)p; p += bytes; return r; };
    float* SDIN = alloc(90112000);   // 64*500*704 f32 (reused as comb_in 16000x512)
    float* A0   = alloc(32768000);   // 32000*256
    float* A1   = alloc(32768000);
    float* Qb   = alloc(32768000);
    float* Kb   = alloc(32768000);
    float* Vb   = alloc(32768000);
    float* ATb  = alloc(32768000);
    float* F1b  = alloc(65536000);   // 32000*512
    float* SCb  = alloc(33554432);   // 8*4*512*512
    Bufs bf = { A0, A1, Qb, Kb, Vb, ATb, F1b, SCb };

    // 1) broadcast-concat feats+anchors -> [64*500, 704]
    long totalC = 32000L * 704;
    concat_kernel<<<(unsigned)((totalC + 255) / 256), 256, 0, stream>>>(
        feats, anchors, SDIN, totalC);

    // 2) det encoder: 64 seqs, 32000 tokens, idim 704 -> final LN in A1
    run_encoder(SDIN, 704, 32000, 64, det, bf, stream);

    // 3) pos-enc + [b,a,t,c] -> [b,t,a*256+c] into SDIN (reused)
    long totalP = 16000L * 512;
    posenc_transpose_kernel<<<(unsigned)((totalP + 255) / 256), 256, 0, stream>>>(
        A1, SDIN, totalP);

    // 4) comb encoder: 32 seqs, 16000 tokens, idim 512 -> final LN in A1
    run_encoder(SDIN, 512, 16000, 32, comb, bf, stream);

    // 5) output head: [32,500,2,128] @ Wout + bout -> [32,500,2]
    out_proj_kernel<<<(32000 + 255) / 256, 256, 0, stream>>>(
        A1, Wout, bout, (float*)d_out, 32000);
}
